// GCN_4337916969193
// MI455X (gfx1250) — matrix operand order repaired and checked by reading, not compiled, not run on hardware
//
#include <hip/hip_runtime.h>
#include <hip/hip_bf16.h>
#include <stdint.h>

// ---------------------------------------------------------------------------
// EGAT 2-layer forward for MI455X (gfx1250, wave32, WMMA).
// Dense projections -> bf16 WMMA 16x16x32 with async-to-LDS A staging and
// LDS-transposed bf16 weights. Edge phase -> scalar, atomics (bandwidth bound).
// ---------------------------------------------------------------------------

#define GN 100000      // nodes
#define GE 1600000     // edges
#define GH 4           // heads
#define SLOPE 0.2f

typedef __attribute__((ext_vector_type(16))) __bf16 v16bf;
typedef __attribute__((ext_vector_type(8)))  __bf16 v8bf;
typedef __attribute__((ext_vector_type(4)))  __bf16 v4bf;
typedef __attribute__((ext_vector_type(8)))  float  v8f;

// ---------------------------------------------------------------------------
// bf16-WMMA GEMM: C[M,Ncols] = A[M,K] @ W[K,Ncols] (+ bias).
// One wave per 16-row x 64-col strip.
//  - A tile (16 x K f32) staged raw into LDS via global_load_async_to_lds_b128
//    (per-lane 16B copies, padded rows to dodge bank conflicts).
//  - W tile (K x 64) staged transposed + converted to bf16 in LDS.
//  - Fragments built from contiguous 16B ds loads; A frag reused over 4 tiles.
// Requires M % 16 == 0, K in {32,64,128}, Ncols % 64 == 0.
// ---------------------------------------------------------------------------
template<int K>
__launch_bounds__(32)
__global__ void gemm_wmma(const float* __restrict__ A,
                          const float* __restrict__ W,
                          const float* __restrict__ bias,
                          float* __restrict__ C,
                          int Ncols) {
  constexpr int SA = K + 4;   // f32 elements per LDS A row   (+16B pad)
  constexpr int SW = K + 8;   // bf16 elements per LDS W col  (+16B pad)
  __shared__ float  lA[16 * SA];
  __shared__ __bf16 lW[64 * SW];

  const int lane = threadIdx.x & 31;
  const int half = lane >> 4;      // 0: lanes 0-15, 1: lanes 16-31
  const int l16  = lane & 15;
  const int m0 = blockIdx.x * 16;
  const int n0 = blockIdx.y * 64;

  // ---- stage A tile: async global->LDS, 16 bytes per lane per issue ----
  {
    const float* abase = A + (size_t)m0 * K;
    const uint32_t lbase = (uint32_t)(uintptr_t)(&lA[0]);  // LDS offset = low 32b
#pragma unroll
    for (int idx = 0; idx < 16 * (K / 4) / 32; ++idx) {
      const int e   = idx * 32 + lane;
      const int row = e / (K / 4);
      const int ch  = (e % (K / 4)) * 4;
      const uint64_t g = (uint64_t)(uintptr_t)(abase + (size_t)row * K + ch);
      const uint32_t l = lbase + (uint32_t)((row * SA + ch) * 4);
      asm volatile("global_load_async_to_lds_b128 %0, %1, off"
                   :: "v"(l), "v"(g) : "memory");
    }
  }

  // ---- stage W tile transposed to bf16: lW[c][k], coalesced global reads ----
  {
#pragma unroll
    for (int idx = 0; idx < 64 * (K / 4) / 32; ++idx) {
      const int e  = idx * 32 + lane;
      const int c  = e & 63;
      const int kq = (e >> 6) * 4;
      const float* wp = W + (size_t)kq * Ncols + n0 + c;
      v4bf pk;
      pk[0] = (__bf16)wp[0];
      pk[1] = (__bf16)wp[(size_t)Ncols];
      pk[2] = (__bf16)wp[(size_t)2 * Ncols];
      pk[3] = (__bf16)wp[(size_t)3 * Ncols];
      *(v4bf*)&lW[c * SW + kq] = pk;
    }
  }

  // wait for async A-tile copies; DS ops from this wave are in-order.
  asm volatile("s_wait_asynccnt 0x0" ::: "memory");

  v8f acc0 = {}, acc1 = {}, acc2 = {}, acc3 = {};
  const float* ar = &lA[l16 * SA];

#pragma unroll
  for (int k0 = 0; k0 < K; k0 += 32) {
    // ---- A fragment: lanes 0-15 K=[k0..k0+15],
    //      lanes 16-31 K=[k0+16..k0+31]; contiguous 16B LDS loads
    const int ka = k0 + half * 16;
    const float4 x0 = *(const float4*)&ar[ka];
    const float4 x1 = *(const float4*)&ar[ka + 4];
    const float4 y0 = *(const float4*)&ar[ka + 8];
    const float4 y1 = *(const float4*)&ar[ka + 12];
    v16bf a;
    a[0]  = (__bf16)x0.x; a[1]  = (__bf16)x0.y; a[2]  = (__bf16)x0.z; a[3]  = (__bf16)x0.w;
    a[4]  = (__bf16)x1.x; a[5]  = (__bf16)x1.y; a[6]  = (__bf16)x1.z; a[7]  = (__bf16)x1.w;
    a[8]  = (__bf16)y0.x; a[9]  = (__bf16)y0.y; a[10] = (__bf16)y0.z; a[11] = (__bf16)y0.w;
    a[12] = (__bf16)y1.x; a[13] = (__bf16)y1.y; a[14] = (__bf16)y1.z; a[15] = (__bf16)y1.w;

    // ---- 4 column tiles sharing the A fragment ----
    const int kb = k0 + half * 16;
#pragma unroll
    for (int t = 0; t < 4; ++t) {
      const __bf16* wr = &lW[(t * 16 + l16) * SW + kb];
      const v8bf b0 = *(const v8bf*)&wr[0];
      const v8bf b1 = *(const v8bf*)&wr[8];
      const v16bf b = __builtin_shufflevector(b0, b1, 0, 1, 2, 3, 4, 5, 6, 7,
                                              8, 9, 10, 11, 12, 13, 14, 15);
      if      (t == 0) acc0 = __builtin_amdgcn_wmma_f32_16x16x32_bf16(false, a, false, b, (short)0, acc0, false, false);
      else if (t == 1) acc1 = __builtin_amdgcn_wmma_f32_16x16x32_bf16(false, a, false, b, (short)0, acc1, false, false);
      else if (t == 2) acc2 = __builtin_amdgcn_wmma_f32_16x16x32_bf16(false, a, false, b, (short)0, acc2, false, false);
      else             acc3 = __builtin_amdgcn_wmma_f32_16x16x32_bf16(false, a, false, b, (short)0, acc3, false, false);
    }
  }

  // ---- store: C-layout VGPR r -> row m0 + r + 8*half, column n0+t*16+l16 ----
#pragma unroll
  for (int t = 0; t < 4; ++t) {
    const int col = n0 + t * 16 + l16;
    const float bv = bias ? bias[col] : 0.0f;
    v8f acc = (t == 0) ? acc0 : (t == 1) ? acc1 : (t == 2) ? acc2 : acc3;
#pragma unroll
    for (int r = 0; r < 8; ++r)
      C[(size_t)(m0 + r + half * 8) * Ncols + col] = acc[r] + bv;
  }
}

// ---------------------------------------------------------------------------
// Utility kernels
// ---------------------------------------------------------------------------
__global__ void fill_f32(float* __restrict__ p, size_t n, float v) {
  size_t i = (size_t)blockIdx.x * blockDim.x + threadIdx.x;
  if (i < n) p[i] = v;
}

__device__ __forceinline__ void atomicMaxF(float* addr, float value) {
  if (value >= 0.0f) atomicMax((int*)addr, __float_as_int(value));
  else               atomicMin((unsigned int*)addr, __float_as_uint(value));
}

// ---------------------------------------------------------------------------
// Layer 1 edge projection: f = LeakyReLU(f_ni[src] + efp + f_nj[dst] + b),
// attn logits per head, atomic running max per (dst, head).
// f is updated in-place in the efp buffer ([E,64]).
// ---------------------------------------------------------------------------
__global__ void egat1_edge(float* __restrict__ f,
                           const float* __restrict__ fni,
                           const float* __restrict__ fnj,
                           const float* __restrict__ bedge,
                           const float* __restrict__ attn,   // [64] = [H,16]
                           const int* __restrict__ src,
                           const int* __restrict__ dst,
                           float* __restrict__ logit,        // [E,4]
                           float* __restrict__ mmax,         // [N,4]
                           int nE) {
  int e = blockIdx.x * blockDim.x + threadIdx.x;
  if (e >= nE) return;
  const int s = src[e], d = dst[e];
  const float* pi = fni + (size_t)s * 64;
  const float* pj = fnj + (size_t)d * 64;
  float* pf = f + (size_t)e * 64;
  float acc[GH] = {0.f, 0.f, 0.f, 0.f};
#pragma unroll
  for (int j = 0; j < 64; ++j) {
    float v = pi[j] + pf[j] + pj[j] + bedge[j];
    v = v > 0.0f ? v : SLOPE * v;
    pf[j] = v;
    acc[j >> 4] += v * attn[j];
  }
#pragma unroll
  for (int h = 0; h < GH; ++h) {
    logit[(size_t)e * GH + h] = acc[h];
    atomicMaxF(&mmax[(size_t)d * GH + h], acc[h]);
  }
}

// ---------------------------------------------------------------------------
// Segment softmax numerator: a = exp(logit - m[dst]); sum into s[dst].
// logits buffer is transformed in place into exp values. One thread per (e,h).
// ---------------------------------------------------------------------------
__global__ void seg_softmax_num(float* __restrict__ a,
                                const int* __restrict__ dst,
                                const float* __restrict__ mmax,
                                float* __restrict__ ssum,
                                int nEH) {
  int t = blockIdx.x * blockDim.x + threadIdx.x;
  if (t >= nEH) return;
  const int e = t >> 2, h = t & 3;
  const int d = dst[e];
  float v = __expf(a[t] - mmax[(size_t)d * GH + h]);
  a[t] = v;
  atomicAdd(&ssum[(size_t)d * GH + h], v);
}

// ---------------------------------------------------------------------------
// Layer 1 message passing: out1[dst,h,:] += ft1[src,h,:] * (a / s[dst,h])
// ---------------------------------------------------------------------------
__global__ void egat1_msg(const float* __restrict__ a,
                          const float* __restrict__ ssum,
                          const float* __restrict__ ft,    // [N, 256]
                          const int* __restrict__ src,
                          const int* __restrict__ dst,
                          float* __restrict__ out,         // [N, 256]
                          int nEH) {
  int t = blockIdx.x * blockDim.x + threadIdx.x;
  if (t >= nEH) return;
  const int e = t >> 2, h = t & 3;
  const int s = src[e], d = dst[e];
  const float sv = ssum[(size_t)d * GH + h];
  const float w = a[t] / (sv == 0.0f ? 1.0f : sv);
  const float* pf = ft + (size_t)s * 256 + h * 64;
  float* po = out + (size_t)d * 256 + h * 64;
#pragma unroll
  for (int j = 0; j < 64; ++j) atomicAdd(&po[j], pf[j] * w);
}

// ---------------------------------------------------------------------------
// relu + mean over heads: out[n,j] = mean_h relu(in[n,h,j]),  D = per-head dim
// ---------------------------------------------------------------------------
__global__ void relu_mean_heads(const float* __restrict__ in,
                                float* __restrict__ out,
                                int nN, int D) {
  int t = blockIdx.x * blockDim.x + threadIdx.x;
  if (t >= nN * D) return;
  const int n = t / D, j = t % D;
  const float* p = in + (size_t)n * GH * D + j;
  float s = 0.f;
#pragma unroll
  for (int h = 0; h < GH; ++h) { float v = p[(size_t)h * D]; s += v > 0.f ? v : 0.f; }
  out[t] = 0.25f * s;
}

// ---------------------------------------------------------------------------
// Layer 2 narrow node projections (4 output cols each — below WMMA tile width)
// ---------------------------------------------------------------------------
__global__ void proj2_ninj(const float* __restrict__ hm,   // [N,64]
                           const float* __restrict__ Wni,  // [64,4]
                           const float* __restrict__ Wnj,  // [64,4]
                           float* __restrict__ ni, float* __restrict__ nj,
                           int nN) {
  int t = blockIdx.x * blockDim.x + threadIdx.x;
  if (t >= nN * GH) return;
  const int n = t >> 2, h = t & 3;
  const float* p = hm + (size_t)n * 64;
  float a = 0.f, b = 0.f;
#pragma unroll
  for (int k = 0; k < 64; ++k) { float x = p[k]; a += x * Wni[k * GH + h]; b += x * Wnj[k * GH + h]; }
  ni[t] = a; nj[t] = b;
}

// ---------------------------------------------------------------------------
// Layer 2 edge projection. fe = relu(f).mean(heads) recomputed from the stored
// layer-1 edge features ([E,64]) to avoid a 100 MB intermediate.
// ---------------------------------------------------------------------------
__global__ void egat2_edge(const float* __restrict__ f,      // [E,64]
                           const float* __restrict__ ni,     // [N,4]
                           const float* __restrict__ nj,     // [N,4]
                           const float* __restrict__ Wfij,   // [16,4]
                           const float* __restrict__ bedge,  // [4]
                           const float* __restrict__ attn,   // [4]
                           const int* __restrict__ src,
                           const int* __restrict__ dst,
                           float* __restrict__ logit,        // [E,4]
                           float* __restrict__ mmax,         // [N,4]
                           int nE) {
  int e = blockIdx.x * blockDim.x + threadIdx.x;
  if (e >= nE) return;
  const int s = src[e], d = dst[e];
  const float* pf = f + (size_t)e * 64;
  float fe[16];
#pragma unroll
  for (int dd = 0; dd < 16; ++dd) {
    float sum = 0.f;
#pragma unroll
    for (int h = 0; h < GH; ++h) { float v = pf[h * 16 + dd]; sum += v > 0.f ? v : 0.f; }
    fe[dd] = 0.25f * sum;
  }
#pragma unroll
  for (int h = 0; h < GH; ++h) {
    float v = ni[(size_t)s * GH + h] + nj[(size_t)d * GH + h] + bedge[h];
#pragma unroll
    for (int dd = 0; dd < 16; ++dd) v += fe[dd] * Wfij[dd * GH + h];
    v = v > 0.f ? v : SLOPE * v;
    const float lg = v * attn[h];
    logit[(size_t)e * GH + h] = lg;
    atomicMaxF(&mmax[(size_t)d * GH + h], lg);
  }
}

// ---------------------------------------------------------------------------
// Layer 2 message passing: out[dst,h,0:16] += ft2[src,h,0:16] * (a/s)
// ---------------------------------------------------------------------------
__global__ void egat2_msg(const float* __restrict__ a,
                          const float* __restrict__ ssum,
                          const float* __restrict__ ft2,   // [N,64]
                          const int* __restrict__ src,
                          const int* __restrict__ dst,
                          float* __restrict__ out,         // [N,64] = [N,4,16]
                          int nEH) {
  int t = blockIdx.x * blockDim.x + threadIdx.x;
  if (t >= nEH) return;
  const int e = t >> 2, h = t & 3;
  const int s = src[e], d = dst[e];
  const float sv = ssum[(size_t)d * GH + h];
  const float w = a[t] / (sv == 0.0f ? 1.0f : sv);
  const float* pf = ft2 + (size_t)s * 64 + h * 16;
  float* po = out + (size_t)d * 64 + h * 16;
#pragma unroll
  for (int c = 0; c < 16; ++c) atomicAdd(&po[c], pf[c] * w);
}

// ---------------------------------------------------------------------------
extern "C" void kernel_launch(void* const* d_in, const int* in_sizes, int n_in,
                              void* d_out, int out_size, void* d_ws, size_t ws_size,
                              hipStream_t stream) {
  const float* nfeat   = (const float*)d_in[0];
  const float* efeat   = (const float*)d_in[1];
  const int*   src     = (const int*)  d_in[2];
  const int*   dst     = (const int*)  d_in[3];
  const float* W1_ni   = (const float*)d_in[4];
  const float* W1_fij  = (const float*)d_in[5];
  const float* W1_nj   = (const float*)d_in[6];
  const float* b1_edge = (const float*)d_in[7];
  const float* W1_node = (const float*)d_in[8];
  const float* b1_node = (const float*)d_in[9];
  const float* attn1   = (const float*)d_in[10];
  const float* W2_ni   = (const float*)d_in[11];
  const float* W2_fij  = (const float*)d_in[12];
  const float* W2_nj   = (const float*)d_in[13];
  const float* b2_edge = (const float*)d_in[14];
  const float* W2_node = (const float*)d_in[15];
  const float* b2_node = (const float*)d_in[16];
  const float* attn2   = (const float*)d_in[17];
  float* out = (float*)d_out;   // [N, H, 16]

  // ---- workspace layout (floats) ----
  float* ws = (float*)d_ws;
  size_t off = 0;
  auto alloc = [&](size_t n) { float* p = ws + off; off += n; return p; };
  float* fbuf  = alloc((size_t)GE * 64);   // efp, then layer-1 edge features f
  float* f_ni  = alloc((size_t)GN * 64);
  float* f_nj  = alloc((size_t)GN * 64);
  float* ft1   = alloc((size_t)GN * 256);
  float* log1  = alloc((size_t)GE * GH);   // logits -> exp numerators (in place)
  float* m1    = alloc((size_t)GN * GH);
  float* s1    = alloc((size_t)GN * GH);
  float* out1  = alloc((size_t)GN * 256);
  float* hmean = alloc((size_t)GN * 64);
  float* ft2   = alloc((size_t)GN * 64);
  float* hni2  = alloc((size_t)GN * GH);
  float* hnj2  = alloc((size_t)GN * GH);
  float* log2  = alloc((size_t)GE * GH);
  float* m2    = alloc((size_t)GN * GH);
  float* s2    = alloc((size_t)GN * GH);
  (void)ws_size; (void)in_sizes; (void)n_in;

  const float NEG_INF = -__builtin_huge_valf();
  auto fill = [&](float* p, size_t n, float v) {
    fill_f32<<<(unsigned)((n + 255) / 256), 256, 0, stream>>>(p, n, v);
  };

  // ---- init accumulators ----
  fill(m1, (size_t)GN * GH, NEG_INF);
  fill(s1, (size_t)GN * GH, 0.0f);
  fill(out1, (size_t)GN * 256, 0.0f);
  fill(m2, (size_t)GN * GH, NEG_INF);
  fill(s2, (size_t)GN * GH, 0.0f);
  fill(out, (size_t)out_size, 0.0f);

  const dim3 wv(32);
  // ---- layer 1 dense projections (WMMA) ----
  gemm_wmma<128><<<dim3(GN / 16, 1), wv, 0, stream>>>(nfeat, W1_ni, nullptr, f_ni, 64);
  gemm_wmma<128><<<dim3(GN / 16, 1), wv, 0, stream>>>(nfeat, W1_nj, nullptr, f_nj, 64);
  gemm_wmma<128><<<dim3(GN / 16, 4), wv, 0, stream>>>(nfeat, W1_node, b1_node, ft1, 256);
  gemm_wmma<32> <<<dim3(GE / 16, 1), wv, 0, stream>>>(efeat, W1_fij, nullptr, fbuf, 64);

  // ---- layer 1 edge phase ----
  const int EB  = (GE + 255) / 256;
  const int EHB = (GE * GH + 255) / 256;
  egat1_edge<<<EB, 256, 0, stream>>>(fbuf, f_ni, f_nj, b1_edge, attn1, src, dst, log1, m1, GE);
  seg_softmax_num<<<EHB, 256, 0, stream>>>(log1, dst, m1, s1, GE * GH);
  egat1_msg<<<EHB, 256, 0, stream>>>(log1, s1, ft1, src, dst, out1, GE * GH);

  // ---- inter-layer relu + head mean ----
  relu_mean_heads<<<(GN * 64 + 255) / 256, 256, 0, stream>>>(out1, hmean, GN, 64);

  // ---- layer 2 dense projections ----
  gemm_wmma<64><<<dim3(GN / 16, 1), wv, 0, stream>>>(hmean, W2_node, b2_node, ft2, 64);
  proj2_ninj<<<(GN * GH + 255) / 256, 256, 0, stream>>>(hmean, W2_ni, W2_nj, hni2, hnj2, GN);

  // ---- layer 2 edge phase ----
  egat2_edge<<<EB, 256, 0, stream>>>(fbuf, hni2, hnj2, W2_fij, b2_edge, attn2, src, dst, log2, m2, GE);
  seg_softmax_num<<<EHB, 256, 0, stream>>>(log2, dst, m2, s2, GE * GH);
  egat2_msg<<<EHB, 256, 0, stream>>>(log2, s2, ft2, src, dst, out, GE * GH);
}
